// SPPF_6700148982068
// MI455X (gfx1250) — matrix operand order repaired and checked by reading, not compiled, not run on hardware
//
#include <hip/hip_runtime.h>
#include <cstdint>
#include <cstddef>

// ---------------------------------------------------------------------------
// Types / helpers
// ---------------------------------------------------------------------------
typedef unsigned short u16t;
typedef __attribute__((ext_vector_type(16))) __bf16 v16bf;
typedef __attribute__((ext_vector_type(8)))  float  v8f;
typedef __attribute__((ext_vector_type(4)))  unsigned int u32x4;
typedef __attribute__((ext_vector_type(8)))  int  i32x8;
typedef __attribute__((ext_vector_type(4)))  int  i32x4;

union Frag  { v16bf v; uint4 q[2]; };
union Pack8 { u16t u[8]; uint4 q; };

__device__ __forceinline__ u16t f2bf(float f) {           // f32 -> bf16 RNE
  unsigned u = __float_as_uint(f);
  return (u16t)((u + 0x7fffu + ((u >> 16) & 1u)) >> 16);
}
__device__ __forceinline__ float bf2f(u16t h) { return __uint_as_float(((unsigned)h) << 16); }

__device__ __forceinline__ v8f wmma_bf16(Frag a, Frag b, v8f c) {
  return __builtin_amdgcn_wmma_f32_16x16x32_bf16(false, a.v, false, b.v, (short)0, c, false, false);
}
__device__ __forceinline__ float silu(float x) { return x / (1.0f + __expf(-x)); }
__device__ __forceinline__ unsigned lds_off(const void* p) { return (unsigned)(size_t)p; }

// ---------------------------------------------------------------------------
// TDM: DMA a (rows x 32) bf16 tile (row pitch = `pitch` elements) from global
// into LDS with 4-DWORD padding every 16 DWORDs -> LDS row pitch = 40 u16 (80B,
// 16B-aligned rows, bank-conflict-free fragment reads).
// ---------------------------------------------------------------------------
#define LDSP 40   // LDS row pitch in u16 elements (32 data + 8 pad)

__device__ __forceinline__ void tdm_load_tile(unsigned lds_addr, const u16t* g,
                                              int rows, unsigned pitch_elems) {
  unsigned long long ga = (unsigned long long)(size_t)g;
  u32x4 g0; i32x8 g1;
  g0[0] = 1u;                                           // count=1 (valid user D#)
  g0[1] = lds_addr;                                     // LDS byte address
  g0[2] = (unsigned)ga;                                 // global addr [31:0]
  g0[3] = (unsigned)((ga >> 32) & 0x01ffffffu) | (2u << 30);  // addr[56:32] | type=2
  unsigned td1 = (unsigned)rows;
  unsigned long long s0 = pitch_elems;
  // data_size=1 (2B) | pad_enable | pad_interval=16dw (code 3) | pad_amount=4dw (code 3)
  g1[0] = (int)((1u << 16) | (1u << 20) | (3u << 22) | (3u << 25));
  g1[1] = (int)((32u & 0xffffu) << 16);                 // tensor_dim0 = 32 (lo16 @63:48)
  g1[2] = (int)(((32u >> 16) & 0xffffu) | ((td1 & 0xffffu) << 16)); // dim0 hi | dim1 lo
  g1[3] = (int)(((td1 >> 16) & 0xffffu) | (32u << 16)); // dim1 hi | tile_dim0 = 32
  g1[4] = (int)(td1 & 0xffffu);                         // tile_dim1 = rows, tile_dim2 = 0
  g1[5] = (int)(unsigned)(s0 & 0xffffffffu);            // tensor_dim0_stride lo32
  g1[6] = (int)(unsigned)((s0 >> 32) & 0xffffu);        // stride hi16 | stride1 lo = 0
  g1[7] = 0;
  i32x4 z4 = (i32x4)0;
#if defined(__clang_major__) && (__clang_major__ >= 23)
  i32x8 z8 = (i32x8)0;
  __builtin_amdgcn_tensor_load_to_lds(g0, g1, z4, z4, z8, 0);
#else
  __builtin_amdgcn_tensor_load_to_lds(g0, g1, z4, z4, 0);
#endif
}

// ---------------------------------------------------------------------------
// Problem constants
// ---------------------------------------------------------------------------
#define BB 8
#define HW 4096
#define C1 512
#define CC 256      // c_ = C1/2
#define CT 1792     // C  = 7*c_
#define CQ 224      // C/8
#define C2 512

// ---------------------------------------------------------------------------
// Weight prep: fold BN scale into bf16 weights; fold bias.
// ---------------------------------------------------------------------------
__global__ __launch_bounds__(256) void k_fold_weight(const float* __restrict__ w,
                                                     const float* __restrict__ g,
                                                     const float* __restrict__ v,
                                                     u16t* __restrict__ out,
                                                     int M, int K) {
  size_t i = (size_t)blockIdx.x * 256 + threadIdx.x;
  if (i >= (size_t)M * K) return;
  int m = (int)(i / K);
  float s = g ? g[m] * rsqrtf(v[m] + 1e-5f) : 1.0f;
  out[i] = f2bf(w[i] * s);
}
__global__ __launch_bounds__(256) void k_fold_bias(const float* __restrict__ bsrc,
                                                   const float* __restrict__ g,
                                                   const float* __restrict__ mn,
                                                   const float* __restrict__ v,
                                                   float* __restrict__ out, int M) {
  int i = blockIdx.x * 256 + threadIdx.x;
  if (i >= M) return;
  float s = g ? g[i] * rsqrtf(v[i] + 1e-5f) : 1.0f;
  out[i] = bsrc[i] - mn[i] * s;
}

// ---------------------------------------------------------------------------
// NCHW f32 -> NHWC bf16 transpose-convert (32x32 LDS tiles, coalesced both ways)
// ---------------------------------------------------------------------------
__global__ __launch_bounds__(256) void k_transpose_bf16(const float* __restrict__ in,
                                                        u16t* __restrict__ out, int Cdim) {
  __shared__ float tile[32][33];
  int b = blockIdx.z, c0 = blockIdx.y * 32, p0 = blockIdx.x * 32;
  const float* src = in + (size_t)b * Cdim * HW;
  u16t* dst = out + (size_t)b * HW * Cdim;
#pragma unroll
  for (int k = 0; k < 4; ++k) {
    int c = c0 + threadIdx.y + k * 8;
    tile[threadIdx.y + k * 8][threadIdx.x] = src[(size_t)c * HW + p0 + threadIdx.x];
  }
  __syncthreads();
#pragma unroll
  for (int k = 0; k < 4; ++k) {
    int p = p0 + threadIdx.y + k * 8;
    dst[(size_t)p * Cdim + c0 + threadIdx.x] = f2bf(tile[threadIdx.x][threadIdx.y + k * 8]);
  }
}

// ---------------------------------------------------------------------------
// Main TN GEMM: D[m,n] = sum_k A[m,k]*Bt[n,k]  (A: [M,K] bf16 weights,
// Bt: [N=4096,K] bf16 activations per batch).  128x128 tile, 8 waves,
// K-step 32, TDM double-buffered LDS with padded pitch.
// EPI: 0 = bias+SiLU -> bf16 [n*ldo+m] ; 1 = bias -> bf16 ; 2 = bias+SiLU -> f32 [m*ldo+n]
// ---------------------------------------------------------------------------
template <int EPI>
__global__ __launch_bounds__(256, 2) void k_gemm_tn(const u16t* __restrict__ A,
                                                    const u16t* __restrict__ Bt,
                                                    const float* __restrict__ bias,
                                                    void* __restrict__ OutP,
                                                    int M, int K, int ldo,
                                                    long long batchB, long long batchO) {
  __shared__ __align__(16) u16t As[2][128 * LDSP];
  __shared__ __align__(16) u16t Bs[2][128 * LDSP];
  const int tid = threadIdx.x, lane = tid & 31, wave = tid >> 5;
  const int wm = wave & 1, wn = wave >> 1;
  const int m0 = blockIdx.x * 128, n0 = blockIdx.y * 128;
  const u16t* Bb = Bt + (size_t)blockIdx.z * batchB;
  const int arows = (M - m0 < 128) ? (M - m0) : 128;
  const int nk = K >> 5;
  const int lrow = lane & 15, hi = lane >> 4;
  const int kb8 = hi * 8, kb16 = hi * 16;

  v8f acc[4][2] = {};

  if (tid < 32) {                       // wave 0 drives the Tensor Data Mover
    tdm_load_tile(lds_off(As[0]), A + (size_t)m0 * K, arows, (unsigned)K);
    tdm_load_tile(lds_off(Bs[0]), Bb + (size_t)n0 * K, 128, (unsigned)K);
    __builtin_amdgcn_s_wait_tensorcnt(0);
  }
  __syncthreads();

  for (int ks = 0; ks < nk; ++ks) {
    const int cur = ks & 1;
    if (tid < 32 && ks + 1 < nk) {      // prefetch next K-slab while computing
      const int nxt = cur ^ 1;
      tdm_load_tile(lds_off(As[nxt]), A + (size_t)m0 * K + (size_t)(ks + 1) * 32, arows, (unsigned)K);
      tdm_load_tile(lds_off(Bs[nxt]), Bb + (size_t)n0 * K + (size_t)(ks + 1) * 32, 128, (unsigned)K);
    }
    const u16t* Ab = As[cur];
    const u16t* Bf = Bs[cur];
    Frag af[4], bf_[2];
#pragma unroll
    for (int i = 0; i < 4; ++i) {
      const u16t* p = Ab + (wm * 64 + i * 16 + lrow) * LDSP;
      af[i].q[0] = *(const uint4*)(p + kb8);
      af[i].q[1] = *(const uint4*)(p + 16 + kb8);
    }
#pragma unroll
    for (int j = 0; j < 2; ++j) {
      const u16t* p = Bf + (wn * 32 + j * 16 + lrow) * LDSP + kb8;
      bf_[j].q[0] = *(const uint4*)(p);
      bf_[j].q[1] = *(const uint4*)(p + 16);
    }
#pragma unroll
    for (int i = 0; i < 4; ++i)
#pragma unroll
      for (int j = 0; j < 2; ++j) acc[i][j] = wmma_bf16(af[i], bf_[j], acc[i][j]);
    if (tid < 32) __builtin_amdgcn_s_wait_tensorcnt(0);
    __syncthreads();
  }

#pragma unroll
  for (int i = 0; i < 4; ++i) {
    const int mbase = m0 + wm * 64 + i * 16 + hi * 8;   // 8 consecutive m per lane
    if (mbase >= M) continue;
#pragma unroll
    for (int j = 0; j < 2; ++j) {
      const int ncol = n0 + wn * 32 + j * 16 + lrow;
      if (EPI == 2) {
        float* Out = (float*)OutP + (size_t)blockIdx.z * batchO;
#pragma unroll
        for (int r = 0; r < 8; ++r) {
          float xv = acc[i][j][r] + bias[mbase + r];
          Out[(size_t)(mbase + r) * ldo + ncol] = silu(xv);
        }
      } else {
        u16t* Out = (u16t*)OutP + (size_t)blockIdx.z * batchO;
        Pack8 pk;
#pragma unroll
        for (int r = 0; r < 8; ++r) {
          float xv = acc[i][j][r] + bias[mbase + r];
          if (EPI == 0) xv = silu(xv);
          pk.u[r] = f2bf(xv);
        }
        *(uint4*)(Out + (size_t)ncol * ldo + mbase) = pk.q;   // 16B store
      }
    }
  }
}

// ---------------------------------------------------------------------------
// Separable pooling (max / avg) on bf16 NHWC cat buffer (ld = CT)
// ---------------------------------------------------------------------------
template <bool MAXP>
__global__ __launch_bounds__(256) void k_pool_h(const u16t* __restrict__ cat, int ch_in,
                                                float* __restrict__ tmp, int kk) {
  const int b = blockIdx.y, hw = blockIdx.x, c = threadIdx.x;
  const int h = hw >> 6, w = hw & 63, p = kk >> 1;
  float acc = MAXP ? -INFINITY : 0.0f;
  for (int d = -p; d <= p; ++d) {
    int ww = w + d;
    if (0 <= ww && ww < 64) {
      float xv = bf2f(cat[((size_t)(b * HW + h * 64 + ww)) * CT + ch_in + c]);
      acc = MAXP ? fmaxf(acc, xv) : acc + xv;
    }
  }
  tmp[((size_t)(b * HW + hw)) * CC + c] = acc;
}
template <bool MAXP>
__global__ __launch_bounds__(256) void k_pool_v(const float* __restrict__ tmp,
                                                u16t* __restrict__ cat, int ch_out,
                                                int kk, float scale) {
  const int b = blockIdx.y, hw = blockIdx.x, c = threadIdx.x;
  const int h = hw >> 6, w = hw & 63, p = kk >> 1;
  float acc = MAXP ? -INFINITY : 0.0f;
  for (int d = -p; d <= p; ++d) {
    int hh = h + d;
    if (0 <= hh && hh < 64) {
      float xv = tmp[((size_t)(b * HW + hh * 64 + w)) * CC + c];
      acc = MAXP ? fmaxf(acc, xv) : acc + xv;
    }
  }
  cat[((size_t)(b * HW + hw)) * CT + ch_out + c] = f2bf(acc * scale);
}

// ---------------------------------------------------------------------------
// Attention scores: eH per (b,w) column slice, eW per (b,h) row slice.
// 64x64 output, K=224 (7 WMMA k-steps); fragments straight from global.
// ---------------------------------------------------------------------------
__global__ __launch_bounds__(128) void k_attn_eH(const u16t* __restrict__ qt,
                                                 const u16t* __restrict__ kt,
                                                 float* __restrict__ e) {
  const int w = blockIdx.x, b = blockIdx.y;
  const int lane = threadIdx.x & 31, wave = threadIdx.x >> 5;
  const int lrow = lane & 15, hi = lane >> 4, kb8 = hi * 8, kb16 = hi * 16;
  const size_t base = (size_t)b * HW;
  const u16t* arow = qt + (base + (size_t)(wave * 16 + lrow) * 64 + w) * CQ;
  v8f acc[4] = {};
  for (int ks = 0; ks < 7; ++ks) {
    Frag a;
    a.q[0] = *(const uint4*)(arow + ks * 32 + kb8);
    a.q[1] = *(const uint4*)(arow + ks * 32 + 16 + kb8);
#pragma unroll
    for (int j = 0; j < 4; ++j) {
      const u16t* brow = kt + (base + (size_t)(j * 16 + lrow) * 64 + w) * CQ + ks * 32 + kb8;
      Frag bf_;
      bf_.q[0] = *(const uint4*)(brow);
      bf_.q[1] = *(const uint4*)(brow + 16);
      acc[j] = wmma_bf16(a, bf_, acc[j]);
    }
  }
#pragma unroll
  for (int j = 0; j < 4; ++j) {
    const int g = j * 16 + lrow;
#pragma unroll
    for (int r = 0; r < 8; ++r) {
      const int h = wave * 16 + hi * 8 + r;
      e[(base + (size_t)h * 64 + w) * 128 + g] = acc[j][r];
    }
  }
}
__global__ __launch_bounds__(128) void k_attn_eW(const u16t* __restrict__ qt,
                                                 const u16t* __restrict__ kt,
                                                 float* __restrict__ e) {
  const int h = blockIdx.x, b = blockIdx.y;
  const int lane = threadIdx.x & 31, wave = threadIdx.x >> 5;
  const int lrow = lane & 15, hi = lane >> 4, kb8 = hi * 8, kb16 = hi * 16;
  const size_t base = (size_t)b * HW;
  const u16t* arow = qt + (base + (size_t)h * 64 + (wave * 16 + lrow)) * CQ;
  v8f acc[4] = {};
  for (int ks = 0; ks < 7; ++ks) {
    Frag a;
    a.q[0] = *(const uint4*)(arow + ks * 32 + kb8);
    a.q[1] = *(const uint4*)(arow + ks * 32 + 16 + kb8);
#pragma unroll
    for (int j = 0; j < 4; ++j) {
      const u16t* brow = kt + (base + (size_t)h * 64 + (j * 16 + lrow)) * CQ + ks * 32 + kb8;
      Frag bf_;
      bf_.q[0] = *(const uint4*)(brow);
      bf_.q[1] = *(const uint4*)(brow + 16);
      acc[j] = wmma_bf16(a, bf_, acc[j]);
    }
  }
#pragma unroll
  for (int j = 0; j < 4; ++j) {
    const int v = j * 16 + lrow;
#pragma unroll
    for (int r = 0; r < 8; ++r) {
      const int ww = wave * 16 + hi * 8 + r;
      e[(base + (size_t)h * 64 + ww) * 128 + 64 + v] = acc[j][r];
    }
  }
}

// ---------------------------------------------------------------------------
// Softmax over 128 (H-attn | W-attn) with diag mask on the H part -> bf16
// ---------------------------------------------------------------------------
__global__ __launch_bounds__(128) void k_softmax(const float* __restrict__ e,
                                                 u16t* __restrict__ att) {
  __shared__ float red[128];
  const int pos = blockIdx.x, t = threadIdx.x;
  const int h = (pos >> 6) & 63;
  float x = e[(size_t)pos * 128 + t];
  if (t == h) x = -INFINITY;           // diag mask (only hits the eH half: t<64)
  red[t] = x; __syncthreads();
  for (int s = 64; s > 0; s >>= 1) { if (t < s) red[t] = fmaxf(red[t], red[t + s]); __syncthreads(); }
  const float mx = red[0]; __syncthreads();
  const float ex = __expf(x - mx);
  red[t] = ex; __syncthreads();
  for (int s = 64; s > 0; s >>= 1) { if (t < s) red[t] += red[t + s]; __syncthreads(); }
  att[(size_t)pos * 128 + t] = f2bf(ex / red[0]);
}

// ---------------------------------------------------------------------------
// outH: per (b,w): D[h,c] = sum_g attH[h,g] * V[g,c]  (K=64 WMMA, LDS-transposed V)
// ---------------------------------------------------------------------------
__global__ __launch_bounds__(128) void k_outH(const u16t* __restrict__ att,
                                              const u16t* __restrict__ vt,
                                              float* __restrict__ tmpH) {
  __shared__ __align__(16) u16t Vt[128 * 72];
  const int c0 = blockIdx.x * 128, w = blockIdx.y, b = blockIdx.z;
  const int tid = threadIdx.x, lane = tid & 31, wave = tid >> 5;
  const size_t base = (size_t)b * HW;
  for (int i = tid; i < 1024; i += 128) {
    const int g = i >> 4, ch = (i & 15) * 8;
    Pack8 d; d.q = *(const uint4*)(vt + (base + (size_t)g * 64 + w) * CT + c0 + ch);
#pragma unroll
    for (int e2 = 0; e2 < 8; ++e2) Vt[(ch + e2) * 72 + g] = d.u[e2];
  }
  __syncthreads();
  const int lrow = lane & 15, hi = lane >> 4, kb8 = hi * 8, kb16 = hi * 16;
  v8f acc[4][2] = {};
#pragma unroll
  for (int ks = 0; ks < 2; ++ks) {
    Frag a[4], bf_[2];
#pragma unroll
    for (int i = 0; i < 4; ++i) {
      const u16t* p = att + (base + (size_t)(i * 16 + lrow) * 64 + w) * 128 + ks * 32;
      a[i].q[0] = *(const uint4*)(p + kb8);
      a[i].q[1] = *(const uint4*)(p + 16 + kb8);
    }
#pragma unroll
    for (int j = 0; j < 2; ++j) {
      const u16t* p = Vt + (wave * 32 + j * 16 + lrow) * 72 + ks * 32 + kb8;
      bf_[j].q[0] = *(const uint4*)(p);
      bf_[j].q[1] = *(const uint4*)(p + 16);
    }
#pragma unroll
    for (int i = 0; i < 4; ++i)
#pragma unroll
      for (int j = 0; j < 2; ++j) acc[i][j] = wmma_bf16(a[i], bf_[j], acc[i][j]);
  }
#pragma unroll
  for (int i = 0; i < 4; ++i)
#pragma unroll
    for (int j = 0; j < 2; ++j) {
      const int c = c0 + wave * 32 + j * 16 + lrow;
#pragma unroll
      for (int r = 0; r < 8; ++r) {
        const int h = i * 16 + hi * 8 + r;
        tmpH[(base + (size_t)h * 64 + w) * CT + c] = acc[i][j][r];
      }
    }
}

// ---------------------------------------------------------------------------
// outW + fuse: attout = bf16( gamma*(outW + outH) + cat )
// ---------------------------------------------------------------------------
__global__ __launch_bounds__(128) void k_outW(const u16t* __restrict__ att,
                                              const u16t* __restrict__ vt,
                                              const float* __restrict__ tmpH,
                                              const u16t* __restrict__ cat,
                                              u16t* __restrict__ attout,
                                              const float* __restrict__ gammap) {
  __shared__ __align__(16) u16t Vt[128 * 72];
  const int c0 = blockIdx.x * 128, h = blockIdx.y, b = blockIdx.z;
  const int tid = threadIdx.x, lane = tid & 31, wave = tid >> 5;
  const size_t base = (size_t)b * HW;
  for (int i = tid; i < 1024; i += 128) {
    const int v = i >> 4, ch = (i & 15) * 8;
    Pack8 d; d.q = *(const uint4*)(vt + (base + (size_t)h * 64 + v) * CT + c0 + ch);
#pragma unroll
    for (int e2 = 0; e2 < 8; ++e2) Vt[(ch + e2) * 72 + v] = d.u[e2];
  }
  __syncthreads();
  const int lrow = lane & 15, hi = lane >> 4, kb8 = hi * 8, kb16 = hi * 16;
  v8f acc[4][2] = {};
#pragma unroll
  for (int ks = 0; ks < 2; ++ks) {
    Frag a[4], bf_[2];
#pragma unroll
    for (int i = 0; i < 4; ++i) {
      const u16t* p = att + (base + (size_t)h * 64 + (i * 16 + lrow)) * 128 + 64 + ks * 32;
      a[i].q[0] = *(const uint4*)(p + kb8);
      a[i].q[1] = *(const uint4*)(p + 16 + kb8);
    }
#pragma unroll
    for (int j = 0; j < 2; ++j) {
      const u16t* p = Vt + (wave * 32 + j * 16 + lrow) * 72 + ks * 32 + kb8;
      bf_[j].q[0] = *(const uint4*)(p);
      bf_[j].q[1] = *(const uint4*)(p + 16);
    }
#pragma unroll
    for (int i = 0; i < 4; ++i)
#pragma unroll
      for (int j = 0; j < 2; ++j) acc[i][j] = wmma_bf16(a[i], bf_[j], acc[i][j]);
  }
  const float gm = gammap[0];
#pragma unroll
  for (int i = 0; i < 4; ++i)
#pragma unroll
    for (int j = 0; j < 2; ++j) {
      const int c = c0 + wave * 32 + j * 16 + lrow;
#pragma unroll
      for (int r = 0; r < 8; ++r) {
        const int ww = i * 16 + hi * 8 + r;
        const size_t idx = (base + (size_t)h * 64 + ww) * CT + c;
        attout[idx] = f2bf(gm * (acc[i][j][r] + tmpH[idx]) + bf2f(cat[idx]));
      }
    }
}

// ---------------------------------------------------------------------------
// Host launcher
// ---------------------------------------------------------------------------
extern "C" void kernel_launch(void* const* d_in, const int* in_sizes, int n_in,
                              void* d_out, int out_size, void* d_ws, size_t ws_size,
                              hipStream_t stream) {
  const float* x     = (const float*)d_in[0];
  const float* cv1_w = (const float*)d_in[1];
  const float* cv1_g = (const float*)d_in[2];
  const float* cv1_b = (const float*)d_in[3];
  const float* cv1_m = (const float*)d_in[4];
  const float* cv1_v = (const float*)d_in[5];
  const float* q_w   = (const float*)d_in[6];
  const float* q_b   = (const float*)d_in[7];
  const float* k_w   = (const float*)d_in[8];
  const float* k_b   = (const float*)d_in[9];
  const float* v_w   = (const float*)d_in[10];
  const float* v_b   = (const float*)d_in[11];
  const float* gamma = (const float*)d_in[12];
  const float* cv2_w = (const float*)d_in[13];
  const float* cv2_g = (const float*)d_in[14];
  const float* cv2_b = (const float*)d_in[15];
  const float* cv2_m = (const float*)d_in[16];
  const float* cv2_v = (const float*)d_in[17];
  (void)in_sizes; (void)n_in; (void)out_size; (void)ws_size;

  // ---- workspace bump allocator (256B aligned) ----
  char* base = (char*)d_ws; size_t off = 0;
  auto alloc = [&](size_t bytes) -> void* {
    off = (off + 255) & ~(size_t)255;
    void* p = base + off; off += bytes; return p;
  };
  u16t*  x_t    = (u16t*) alloc((size_t)BB * HW * C1 * 2);   // NHWC bf16 input
  u16t*  cat_t  = (u16t*) alloc((size_t)BB * HW * CT * 2);   // fused SPPF concat
  u16t*  q_t    = (u16t*) alloc((size_t)BB * HW * CQ * 2);
  u16t*  k_t    = (u16t*) alloc((size_t)BB * HW * CQ * 2);
  u16t*  v_t    = (u16t*) alloc((size_t)BB * HW * CT * 2);
  float* e_all  = (float*)alloc((size_t)BB * HW * 128 * 4);
  u16t*  att_b  = (u16t*) alloc((size_t)BB * HW * 128 * 2);
  float* tmpH   = (float*)alloc((size_t)BB * HW * CT * 4);
  u16t*  attout = (u16t*) alloc((size_t)BB * HW * CT * 2);
  float* ptmp   = (float*)alloc((size_t)BB * HW * CC * 4);
  u16t*  w1b    = (u16t*) alloc((size_t)CC * C1 * 2);
  u16t*  wqb    = (u16t*) alloc((size_t)CQ * CT * 2);
  u16t*  wkb    = (u16t*) alloc((size_t)CQ * CT * 2);
  u16t*  wvb    = (u16t*) alloc((size_t)CT * CT * 2);
  u16t*  w2b    = (u16t*) alloc((size_t)C2 * CT * 2);
  float* bias1  = (float*)alloc((size_t)CC * 4);
  float* bias2  = (float*)alloc((size_t)C2 * 4);

  // ---- weight prep (BN folded) ----
  k_fold_weight<<<(CC * C1 + 255) / 256, 256, 0, stream>>>(cv1_w, cv1_g, cv1_v, w1b, CC, C1);
  k_fold_bias  <<<(CC + 255) / 256, 256, 0, stream>>>(cv1_b, cv1_g, cv1_m, cv1_v, bias1, CC);
  k_fold_weight<<<(CQ * CT + 255) / 256, 256, 0, stream>>>(q_w, nullptr, nullptr, wqb, CQ, CT);
  k_fold_weight<<<(CQ * CT + 255) / 256, 256, 0, stream>>>(k_w, nullptr, nullptr, wkb, CQ, CT);
  k_fold_weight<<<(CT * CT + 255) / 256, 256, 0, stream>>>(v_w, nullptr, nullptr, wvb, CT, CT);
  k_fold_weight<<<(C2 * CT + 255) / 256, 256, 0, stream>>>(cv2_w, cv2_g, cv2_v, w2b, C2, CT);
  k_fold_bias  <<<(C2 + 255) / 256, 256, 0, stream>>>(cv2_b, cv2_g, cv2_m, cv2_v, bias2, C2);

  // ---- input to NHWC bf16 ----
  k_transpose_bf16<<<dim3(HW / 32, C1 / 32, BB), dim3(32, 8), 0, stream>>>(x, x_t, C1);

  // ---- cv1 (BN+SiLU fused) -> cat channels [0,256) ----
  k_gemm_tn<0><<<dim3(2, 32, BB), 256, 0, stream>>>(w1b, x_t, bias1, cat_t,
      CC, C1, CT, (long long)HW * C1, (long long)HW * CT);

  // ---- SPPF pools (separable: k^2 taps -> 2k taps) into cat channels ----
  dim3 pg(HW, BB);
  k_pool_h<true ><<<pg, 256, 0, stream>>>(cat_t,      0, ptmp, 5);
  k_pool_v<true ><<<pg, 256, 0, stream>>>(ptmp, cat_t, CC * 1, 5, 1.0f);           // y1
  k_pool_h<true ><<<pg, 256, 0, stream>>>(cat_t, CC * 1, ptmp, 5);
  k_pool_v<true ><<<pg, 256, 0, stream>>>(ptmp, cat_t, CC * 2, 5, 1.0f);           // y2
  k_pool_h<true ><<<pg, 256, 0, stream>>>(cat_t, CC * 2, ptmp, 5);
  k_pool_v<true ><<<pg, 256, 0, stream>>>(ptmp, cat_t, CC * 3, 5, 1.0f);           // y3
  k_pool_h<false><<<pg, 256, 0, stream>>>(cat_t,      0, ptmp, 13);
  k_pool_v<false><<<pg, 256, 0, stream>>>(ptmp, cat_t, CC * 4, 13, 1.0f / 169.0f); // a1
  k_pool_h<false><<<pg, 256, 0, stream>>>(cat_t,      0, ptmp, 9);
  k_pool_v<false><<<pg, 256, 0, stream>>>(ptmp, cat_t, CC * 5, 9, 1.0f / 81.0f);   // a2
  k_pool_h<false><<<pg, 256, 0, stream>>>(cat_t,      0, ptmp, 5);
  k_pool_v<false><<<pg, 256, 0, stream>>>(ptmp, cat_t, CC * 6, 5, 1.0f / 25.0f);   // a3

  // ---- q, k, v projections (cat is L2-resident: 117 MB < 192 MB) ----
  k_gemm_tn<1><<<dim3(2, 32, BB), 256, 0, stream>>>(wqb, cat_t, q_b, q_t,
      CQ, CT, CQ, (long long)HW * CT, (long long)HW * CQ);
  k_gemm_tn<1><<<dim3(2, 32, BB), 256, 0, stream>>>(wkb, cat_t, k_b, k_t,
      CQ, CT, CQ, (long long)HW * CT, (long long)HW * CQ);
  k_gemm_tn<1><<<dim3(14, 32, BB), 256, 0, stream>>>(wvb, cat_t, v_b, v_t,
      CT, CT, CT, (long long)HW * CT, (long long)HW * CT);

  // ---- criss-cross attention ----
  k_attn_eH<<<dim3(64, BB), 128, 0, stream>>>(q_t, k_t, e_all);
  k_attn_eW<<<dim3(64, BB), 128, 0, stream>>>(q_t, k_t, e_all);
  k_softmax<<<BB * HW, 128, 0, stream>>>(e_all, att_b);
  k_outH<<<dim3(14, 64, BB), 128, 0, stream>>>(att_b, v_t, tmpH);
  k_outW<<<dim3(14, 64, BB), 128, 0, stream>>>(att_b, v_t, tmpH, cat_t, attout, gamma);

  // ---- cv2 (BN+SiLU fused) -> f32 NCHW output ----
  k_gemm_tn<2><<<dim3(4, 32, BB), 256, 0, stream>>>(w2b, attout, bias2, d_out,
      C2, CT, HW, (long long)HW * CT, (long long)C2 * HW);
}
